// GraphVAE_23845658427756
// MI455X (gfx1250) — matrix operand order repaired and checked by reading, not compiled, not run on hardware
//
#include <hip/hip_runtime.h>

typedef __attribute__((ext_vector_type(16))) _Float16 v16h;
typedef __attribute__((ext_vector_type(8)))  float    v8f;

// A-fragment source modes
#define AMODE_PLAIN    0   // a = A[idx]
#define AMODE_BIASRELU 1   // a = relu(A[idx] + Ab[k])          (h from raw agg)
#define AMODE_REPARAM  2   // a = A[idx] + A3[idx]*exp(A2[idx]) (z from mu,lv,eps)
// Epilogue modes
#define EMODE_PLAIN    0   // out = acc
#define EMODE_HSCALE   1   // out = acc*dinv[m] (=h^), out2 = acc*dinv[m]^2 (agg init)
#define EMODE_SIGMOID  2   // out = sigmoid(acc + bias[col])

// ---------------------------------------------------------------------------
// Degree / normalization
// ---------------------------------------------------------------------------
__global__ void k_fill1(float* __restrict__ p, int n) {
    int t = blockIdx.x * blockDim.x + threadIdx.x;
    if (t < n) p[t] = 1.0f;                       // +1 self-loop
}

__global__ void k_degcnt(const int* __restrict__ ei, float* __restrict__ deg, int E) {
    int t = blockIdx.x * blockDim.x + threadIdx.x;
    if (t < E) unsafeAtomicAdd(&deg[ei[E + t]], 1.0f);   // dst row of edge_index
}

__global__ void k_rsqrt(float* __restrict__ p, int n) {
    int t = blockIdx.x * blockDim.x + threadIdx.x;
    if (t < n) p[t] = rsqrtf(p[t]);
}

// ---------------------------------------------------------------------------
// WMMA GEMM with fused prologue/epilogue.
// out[M,NOUT] = f(A)[M,K] * W[K,NOUT]; f16 WMMA inputs, fp32 accumulate.
// One wave per 16-row strip, all N-tiles; W staged transposed in LDS so each
// B fragment is a contiguous 32B (2x ds_load_b128) read.
// ---------------------------------------------------------------------------
template <int K, int NOUT, int AMODE, int EMODE>
__global__ __launch_bounds__(256)
void k_gemm(const float* __restrict__ A,  const float* __restrict__ A2,
            const float* __restrict__ A3, const float* __restrict__ Ab,
            const float* __restrict__ W,  const float* __restrict__ bias,
            const float* __restrict__ dinv,
            float* __restrict__ out, float* __restrict__ out2, int Mtiles) {
    __shared__ __attribute__((aligned(64))) _Float16 sWT[NOUT * K]; // [n][k]
    __shared__ float sB[K];
    const int tid = threadIdx.x;
    for (int idx = tid; idx < NOUT * K; idx += 256) {
        int n = idx / K, k = idx - n * K;
        sWT[idx] = (_Float16)W[k * NOUT + n];
    }
    if (AMODE == AMODE_BIASRELU)
        for (int idx = tid; idx < K; idx += 256) sB[idx] = Ab[idx];
    __syncthreads();

    const int wave = tid >> 5;
    const int lane = tid & 31;
    const int tile = blockIdx.x * 8 + wave;
    if (tile >= Mtiles) return;                   // wave-uniform: EXEC stays full

    const int ln = lane & 15;
    const int hi = lane >> 4;                     // 0: lanes 0-15, 1: lanes 16-31
    const long rowbase = (long)(tile * 16 + ln) * K;

    v8f zero = {};
    v8f acc[NOUT / 16];
#pragma unroll
    for (int nt = 0; nt < NOUT / 16; ++nt) acc[nt] = zero;

#pragma unroll
    for (int kb = 0; kb < K; kb += 32) {
        // A fragment: lanes<16 hold K {0..15}, lanes>=16 hold {16..31}
        v16h a;
        const int g0 = kb + hi * 16;
        const int g1 = kb + hi * 16 + 8;
#pragma unroll
        for (int j = 0; j < 16; ++j) {
            const int k   = (j < 8 ? g0 + j : g1 + (j - 8));
            const long ix = rowbase + k;
            float v;
            if (AMODE == AMODE_PLAIN)         v = A[ix];
            else if (AMODE == AMODE_BIASRELU) v = fmaxf(A[ix] + sB[k], 0.0f);
            else /* AMODE_REPARAM */          v = A[ix] + A3[ix] * __expf(A2[ix]);
            a[j] = (_Float16)v;
        }

        const int kbase = kb + hi * 16;
#pragma unroll
        for (int nt = 0; nt < NOUT / 16; ++nt) {
            const int n = nt * 16 + ln;
            v16h b = *(const v16h*)&sWT[n * K + kbase];   // 32B contiguous LDS read
            acc[nt] = __builtin_amdgcn_wmma_f32_16x16x32_f16(
                false, a, false, b, (short)0, acc[nt], false, false);
        }
    }

    // C/D layout: VGPR r -> M = r + 8*hi, N = lane&15
#pragma unroll
    for (int r = 0; r < 8; ++r) {
        const int m = tile * 16 + hi * 8 + r;
        float w0 = 0.0f;
        if (EMODE == EMODE_HSCALE) w0 = dinv[m];
#pragma unroll
        for (int nt = 0; nt < NOUT / 16; ++nt) {
            const int col = nt * 16 + ln;
            float v = acc[nt][r];
            if (EMODE == EMODE_PLAIN) {
                out[(long)m * NOUT + col] = v;
            } else if (EMODE == EMODE_HSCALE) {
                float hs = v * w0;                        // h^ = h * dinv[src]
                out [(long)m * NOUT + col] = hs;          // message source
                out2[(long)m * NOUT + col] = hs * w0;     // self-loop agg init
            } else { /* EMODE_SIGMOID */
                v += bias[col];
                out[(long)m * NOUT + col] = 1.0f / (1.0f + __expf(-v));
            }
        }
    }
}

// ---------------------------------------------------------------------------
// agg[dst,:] += h^[src,:] * dinv[dst]   (scatter via f32 HW atomics)
// 16 threads per edge, float4 each; messages pre-scaled by dinv[src] in GEMM.
// ---------------------------------------------------------------------------
__global__ void k_edge_agg(const float* __restrict__ hs, const float* __restrict__ dinv,
                           const int* __restrict__ ei, float* __restrict__ agg, int E) {
    int t = blockIdx.x * blockDim.x + threadIdx.x;
    int e = t >> 4;
    if (e >= E) return;
    int c = (t & 15) << 2;
    int s = ei[e], d = ei[E + e];
    float nd = dinv[d];
    float4 hv = *(const float4*)&hs[(long)s * 64 + c];
    float* ap = &agg[(long)d * 64 + c];
    unsafeAtomicAdd(ap + 0, hv.x * nd);
    unsafeAtomicAdd(ap + 1, hv.y * nd);
    unsafeAtomicAdd(ap + 2, hv.z * nd);
    unsafeAtomicAdd(ap + 3, hv.w * nd);
}

// ---------------------------------------------------------------------------
// io += bias (broadcast over rows of 64)
// ---------------------------------------------------------------------------
__global__ void k_bias64(float* __restrict__ io, const float* __restrict__ b, int total) {
    int t = blockIdx.x * blockDim.x + threadIdx.x;
    if (t >= total) return;
    io[t] = io[t] + b[t & 63];
}

// ---------------------------------------------------------------------------
extern "C" void kernel_launch(void* const* d_in, const int* in_sizes, int n_in,
                              void* d_out, int out_size, void* d_ws, size_t ws_size,
                              hipStream_t stream) {
    const float* x    = (const float*)d_in[0];
    const int*   ei   = (const int*)  d_in[1];
    const float* eps  = (const float*)d_in[2];
    const float* W1   = (const float*)d_in[3];
    const float* b1   = (const float*)d_in[4];
    const float* Wmu  = (const float*)d_in[5];
    const float* bmu  = (const float*)d_in[6];
    const float* Wlv  = (const float*)d_in[7];
    const float* blv  = (const float*)d_in[8];
    const float* Wdec = (const float*)d_in[9];
    const float* bdec = (const float*)d_in[10];

    const int Nn = in_sizes[0] / 128;   // 100000
    const int E  = in_sizes[1] / 2;     // 1600000

    float* recon = (float*)d_out;                     // [N,128]
    float* mu    = recon + (long)Nn * 128;            // [N,64]
    float* lv    = mu    + (long)Nn * 64;             // [N,64]

    float* ws   = (float*)d_ws;
    long   P    = ((long)Nn + 63) & ~63L;
    float* dinv = ws;                                 // [N]
    float* t0   = ws + P;                             // [N,64] h^ (scaled messages)
    float* t1   = t0 + (long)Nn * 64;                 // [N,64] agg1 (raw, pre-bias)
    float* t2   = t1 + (long)Nn * 64;                 // [N,64] h^ for lv conv

    const dim3 B(256);
    const int gN   = (Nn + 255) / 256;
    const int gE   = (E + 255) / 256;
    const int gE16 = (int)(((long)E * 16 + 255) / 256);
    const int gN64 = (Nn * 64 + 255) / 256;
    const int Mt   = (Nn + 15) / 16;
    const int gG   = (Mt + 7) / 8;

    // normalization: dinv = rsqrt(in_degree + 1)
    k_fill1 <<<gN, B, 0, stream>>>(dinv, Nn);
    k_degcnt<<<gE, B, 0, stream>>>(ei, dinv, E);
    k_rsqrt <<<gN, B, 0, stream>>>(dinv, Nn);

    // conv1: t0 = (x@W1)*dinv, t1 = (x@W1)*dinv^2 (agg init); then edge scatter.
    // bias+ReLU of conv1 is fused into the A-loaders of the mu/lv GEMMs.
    k_gemm<128, 64, AMODE_PLAIN, EMODE_HSCALE><<<gG, B, 0, stream>>>(
        x, nullptr, nullptr, nullptr, W1, nullptr, dinv, t0, t1, Mt);
    k_edge_agg<<<gE16, B, 0, stream>>>(t0, dinv, ei, t1, E);

    // mu conv: A = relu(t1 + b1); agg init written straight into d_out's mu block
    k_gemm<64, 64, AMODE_BIASRELU, EMODE_HSCALE><<<gG, B, 0, stream>>>(
        t1, nullptr, nullptr, b1, Wmu, nullptr, dinv, t0, mu, Mt);
    k_edge_agg<<<gE16, B, 0, stream>>>(t0, dinv, ei, mu, E);
    k_bias64<<<gN64, B, 0, stream>>>(mu, bmu, Nn * 64);

    // logvar conv
    k_gemm<64, 64, AMODE_BIASRELU, EMODE_HSCALE><<<gG, B, 0, stream>>>(
        t1, nullptr, nullptr, b1, Wlv, nullptr, dinv, t2, lv, Mt);
    k_edge_agg<<<gE16, B, 0, stream>>>(t2, dinv, ei, lv, E);
    k_bias64<<<gN64, B, 0, stream>>>(lv, blv, Nn * 64);

    // decoder with fused reparameterization: A = mu + eps*exp(lv)
    k_gemm<64, 128, AMODE_REPARAM, EMODE_SIGMOID><<<gG, B, 0, stream>>>(
        mu, lv, eps, nullptr, Wdec, bdec, nullptr, recon, nullptr, Mt);
}
